// PathGNNSynthetic_78615081386031
// MI455X (gfx1250) — compile-verified, matrix operand order checked by reading, not run on hardware
//
#include <hip/hip_runtime.h>
#include <hip/hip_bf16.h>
#include <math.h>

#define N_NODESC   50000
#define N_PATHSC   100000
#define PATH_LENC  5
#define LSEQ       6          // PATH_LEN + 1
#define EMBC       128
#define IN_DIMC    32
#define NUM_GRAPHSC 512
#define NUM_CLASSESC 10
#define NR_LSTMSC  2
#define GATESC     512        // 4*EMB

typedef __attribute__((ext_vector_type(16))) _Float16 v16h;
typedef __attribute__((ext_vector_type(8)))  _Float16 v8h;
typedef __attribute__((ext_vector_type(8)))  float    v8f;

__device__ __forceinline__ v16h make_frag(const _Float16* p0, const _Float16* p1) {
    // A/B 16-bit fragment: VGPR0-3 <- 8 halves at p0, VGPR4-7 <- 8 halves at p1
    v16h a;
    *reinterpret_cast<v8h*>(&a)       = *reinterpret_cast<const v8h*>(p0);
    *(reinterpret_cast<v8h*>(&a) + 1) = *reinterpret_cast<const v8h*>(p1);
    return a;
}

__device__ __forceinline__ float sigmf(float x) { return 1.0f / (1.0f + __expf(-x)); }

// ---------------------------------------------------------------- weight prep
__global__ void k_prep(const float* __restrict__ Wih, const float* __restrict__ Whh,
                       const float* __restrict__ dist_emb,
                       _Float16* __restrict__ Wih_h, float* __restrict__ Wih_last,
                       _Float16* __restrict__ Whh_h, _Float16* __restrict__ dist_h) {
    int id = blockIdx.x * blockDim.x + threadIdx.x;
    const int totalWih = NR_LSTMSC * GATESC * (2 * EMBC + 1);   // [2,512,257]
    if (id < totalWih) {
        int k  = id % 257;
        int rg = id / 257;                                      // layer*512 + gate
        float w = Wih[id];
        if (k == 256) Wih_last[rg] = w;                         // neighbor-mask column
        else          Wih_h[(size_t)rg * 256 + k] = (_Float16)w;
    }
    if (id < NR_LSTMSC * GATESC * EMBC) Whh_h[id] = (_Float16)Whh[id];
    if (id < (PATH_LENC + 2) * EMBC)    dist_h[id] = (_Float16)dist_emb[id];
}

// ---------------------------------------------------------------- node encoder
__global__ void k_encoder(const float* __restrict__ x, const float* __restrict__ W,
                          const float* __restrict__ b, float* __restrict__ h) {
    __shared__ float sx[IN_DIMC];
    int n = blockIdx.x, j = threadIdx.x;
    if (j < IN_DIMC) sx[j] = x[(size_t)n * IN_DIMC + j];
    __syncthreads();
    float acc = b[j];
#pragma unroll
    for (int k = 0; k < IN_DIMC; ++k) acc += sx[k] * W[k * EMBC + j];
    h[(size_t)n * EMBC + j] = acc;
}

// ---------------------------------------------------------------- row L2 normalize
__global__ void k_normalize(const float* __restrict__ h, float* __restrict__ hn,
                            _Float16* __restrict__ hn_h) {
    __shared__ float red[EMBC];
    int n = blockIdx.x, j = threadIdx.x;
    float v = h[(size_t)n * EMBC + j];
    red[j] = v * v;
    __syncthreads();
    for (int s = 64; s > 0; s >>= 1) {
        if (j < s) red[j] += red[j + s];
        __syncthreads();
    }
    float inv = 1.0f / fmaxf(sqrtf(red[0]), 1e-12f);
    float o = v * inv;
    hn[(size_t)n * EMBC + j]   = o;
    hn_h[(size_t)n * EMBC + j] = (_Float16)o;
}

// ---------------------------------------------------------------- fused LSTM step
// One block: 16 paths x 512 gates; 8 waves, each owns 4 16x16 output tiles.
// WMMA K-loop: feat (gathered hn + dist, K=256) and prev-h (K=128), then
// in-LDS LSTM pointwise cell + online "last hidden" capture.
__global__ __launch_bounds__(256) void k_lstm_step(
    int t, int first,
    const int* __restrict__ path_index, const int* __restrict__ distances,
    const float* __restrict__ nmask, const int* __restrict__ path_lengths,
    const _Float16* __restrict__ hn_h, const _Float16* __restrict__ dist_h,
    const _Float16* __restrict__ Wih_h, const float* __restrict__ Wih_last,
    const _Float16* __restrict__ Whh_h,
    const float* __restrict__ bih, const float* __restrict__ bhh,
    _Float16* __restrict__ hs_h, float* __restrict__ cs, float* __restrict__ last) {

    __shared__ __align__(16) _Float16 sA[16][2 * EMBC];   // gathered feat rows (8KB)
    __shared__ __align__(16) _Float16 sH[16][EMBC];       // prev hidden rows (4KB)
    __shared__ float sG[16][GATESC];                      // gate tile (32KB)
    __shared__ int   sNode[16], sDist[16], sLen[16];
    __shared__ float sMask[16];

    const int tid = threadIdx.x;
    const int p0  = blockIdx.x * 16;

    if (tid < 16) {
        int p = p0 + tid;
        sNode[tid] = path_index[(size_t)t * N_PATHSC + p];
        sDist[tid] = distances[(size_t)p * LSEQ + t];
        sMask[tid] = nmask[(size_t)p * LSEQ + t];
        sLen[tid]  = path_lengths[p];
    }
    __syncthreads();

    // stage feat: 16 rows x 256 halves = 512 x 16B chunks
    for (int c = tid; c < 512; c += 256) {
        int m = c >> 5, q = c & 31;
        const _Float16* src = (q < 16)
            ? (hn_h   + (size_t)sNode[m] * EMBC + q * 8)
            : (dist_h + (size_t)sDist[m] * EMBC + (q - 16) * 8);
        *reinterpret_cast<v8h*>(&sA[m][q * 8]) = *reinterpret_cast<const v8h*>(src);
    }
    if (!first) {  // stage prev hidden: 16 rows x 128 halves = 256 x 16B chunks
        int m = tid >> 4, q = tid & 15;
        *reinterpret_cast<v8h*>(&sH[m][q * 8]) =
            *reinterpret_cast<const v8h*>(hs_h + (size_t)(p0 + m) * EMBC + q * 8);
    }
    __syncthreads();

    const int wave = tid >> 5;
    const int lane = tid & 31;
    const int rowc = lane & 15;   // A row M / B col N
    const int sel  = lane >> 4;   // K-half selector
    const int gbase = wave * 64;  // this wave's 64-gate slab

    v8f acc0 = {}, acc1 = {}, acc2 = {}, acc3 = {};

#pragma unroll
    for (int kc = 0; kc < 8; ++kc) {           // feat @ Wih[:, :256].T
        int kb = kc * 32;
        v16h a = make_frag(&sA[rowc][kb + sel * 8], &sA[rowc][kb + 16 + sel * 8]);
        const _Float16* w0 = Wih_h + (size_t)(gbase      + rowc) * 256 + kb;
        const _Float16* w1 = Wih_h + (size_t)(gbase + 16 + rowc) * 256 + kb;
        const _Float16* w2 = Wih_h + (size_t)(gbase + 32 + rowc) * 256 + kb;
        const _Float16* w3 = Wih_h + (size_t)(gbase + 48 + rowc) * 256 + kb;
        v16h b0 = make_frag(w0 + sel * 8, w0 + 16 + sel * 8);
        v16h b1 = make_frag(w1 + sel * 8, w1 + 16 + sel * 8);
        v16h b2 = make_frag(w2 + sel * 8, w2 + 16 + sel * 8);
        v16h b3 = make_frag(w3 + sel * 8, w3 + 16 + sel * 8);
        acc0 = __builtin_amdgcn_wmma_f32_16x16x32_f16(false, a, false, b0, (short)0, acc0, false, false);
        acc1 = __builtin_amdgcn_wmma_f32_16x16x32_f16(false, a, false, b1, (short)0, acc1, false, false);
        acc2 = __builtin_amdgcn_wmma_f32_16x16x32_f16(false, a, false, b2, (short)0, acc2, false, false);
        acc3 = __builtin_amdgcn_wmma_f32_16x16x32_f16(false, a, false, b3, (short)0, acc3, false, false);
    }
    if (!first) {
#pragma unroll
        for (int kc = 0; kc < 4; ++kc) {       // h_prev @ Whh.T
            int kb = kc * 32;
            v16h a = make_frag(&sH[rowc][kb + sel * 8], &sH[rowc][kb + 16 + sel * 8]);
            const _Float16* w0 = Whh_h + (size_t)(gbase      + rowc) * EMBC + kb;
            const _Float16* w1 = Whh_h + (size_t)(gbase + 16 + rowc) * EMBC + kb;
            const _Float16* w2 = Whh_h + (size_t)(gbase + 32 + rowc) * EMBC + kb;
            const _Float16* w3 = Whh_h + (size_t)(gbase + 48 + rowc) * EMBC + kb;
            v16h b0 = make_frag(w0 + sel * 8, w0 + 16 + sel * 8);
            v16h b1 = make_frag(w1 + sel * 8, w1 + 16 + sel * 8);
            v16h b2 = make_frag(w2 + sel * 8, w2 + 16 + sel * 8);
            v16h b3 = make_frag(w3 + sel * 8, w3 + 16 + sel * 8);
            acc0 = __builtin_amdgcn_wmma_f32_16x16x32_f16(false, a, false, b0, (short)0, acc0, false, false);
            acc1 = __builtin_amdgcn_wmma_f32_16x16x32_f16(false, a, false, b1, (short)0, acc1, false, false);
            acc2 = __builtin_amdgcn_wmma_f32_16x16x32_f16(false, a, false, b2, (short)0, acc2, false, false);
            acc3 = __builtin_amdgcn_wmma_f32_16x16x32_f16(false, a, false, b3, (short)0, acc3, false, false);
        }
    }

    // epilogue: D layout -> (M = r + 8*sel, N = rowc); add biases + nmask rank-1 term
#pragma unroll
    for (int r = 0; r < 8; ++r) {
        int m = r + 8 * sel;
        float nm = sMask[m];
        int g0 = gbase + rowc;
        sG[m][g0     ] = acc0[r] + bih[g0     ] + bhh[g0     ] + nm * Wih_last[g0     ];
        sG[m][g0 + 16] = acc1[r] + bih[g0 + 16] + bhh[g0 + 16] + nm * Wih_last[g0 + 16];
        sG[m][g0 + 32] = acc2[r] + bih[g0 + 32] + bhh[g0 + 32] + nm * Wih_last[g0 + 32];
        sG[m][g0 + 48] = acc3[r] + bih[g0 + 48] + bhh[g0 + 48] + nm * Wih_last[g0 + 48];
    }
    __syncthreads();

    // pointwise LSTM cell (torch gate order i,f,g,o), fused last-hidden capture
    for (int e = tid; e < 16 * EMBC; e += 256) {
        int m = e >> 7, j = e & 127;
        size_t pj = (size_t)(p0 + m) * EMBC + j;
        float gi = sG[m][j], gf = sG[m][EMBC + j];
        float gg = sG[m][2 * EMBC + j], go = sG[m][3 * EMBC + j];
        float cold = first ? 0.0f : cs[pj];
        float cnew = sigmf(gf) * cold + sigmf(gi) * tanhf(gg);
        float hval = sigmf(go) * tanhf(cnew);
        cs[pj]   = cnew;
        hs_h[pj] = (_Float16)hval;
        if (sLen[m] - 1 == t) last[pj] = hval;
    }
}

// ---------------------------------------------------------------- utility kernels
__global__ void k_zero(float* __restrict__ p, int n) {
    int id = blockIdx.x * blockDim.x + threadIdx.x;
    if (id < n) p[id] = 0.0f;
}

__global__ void k_scatter_paths(const float* __restrict__ last,
                                const int* __restrict__ mask_index,
                                float* __restrict__ z) {
    int id = blockIdx.x * blockDim.x + threadIdx.x;
    if (id < N_PATHSC * EMBC) {
        int p = id >> 7, j = id & 127;
        atomicAdd(&z[(size_t)mask_index[p] * EMBC + j], last[id]);
    }
}

__global__ void k_bn_stats(const float* __restrict__ z, const float* __restrict__ hn,
                           float* __restrict__ stats) {
    __shared__ float s1[256], s2[256];
    int j = blockIdx.x, tid = threadIdx.x;
    float a = 0.0f, b = 0.0f;
    for (int n = tid; n < N_NODESC; n += 256) {
        float v = z[(size_t)n * EMBC + j] + hn[(size_t)n * EMBC + j];
        a += v; b += v * v;
    }
    s1[tid] = a; s2[tid] = b;
    __syncthreads();
    for (int s = 128; s > 0; s >>= 1) {
        if (tid < s) { s1[tid] += s1[tid + s]; s2[tid] += s2[tid + s]; }
        __syncthreads();
    }
    if (tid == 0) {
        float mean = s1[0] / (float)N_NODESC;
        float var  = s2[0] / (float)N_NODESC - mean * mean;
        stats[j]        = mean;
        stats[EMBC + j] = rsqrtf(var + 1e-5f);
    }
}

__global__ void k_bn_apply(float* __restrict__ z, const float* __restrict__ hn,
                           const float* __restrict__ stats,
                           const float* __restrict__ gamma, const float* __restrict__ beta) {
    int id = blockIdx.x * blockDim.x + threadIdx.x;
    if (id < N_NODESC * EMBC) {
        int j = id & 127;
        float v = z[id] + hn[id];
        z[id] = (v - stats[j]) * stats[EMBC + j] * gamma[j] + beta[j];
    }
}

__global__ void k_scatter_graph(const float* __restrict__ h, const int* __restrict__ batch,
                                float* __restrict__ g) {
    int id = blockIdx.x * blockDim.x + threadIdx.x;
    if (id < N_NODESC * EMBC) {
        int n = id >> 7, j = id & 127;
        atomicAdd(&g[(size_t)batch[n] * EMBC + j], h[id]);
    }
}

__global__ void k_mlp(const float* __restrict__ g, const float* __restrict__ W1,
                      const float* __restrict__ b1, const float* __restrict__ W2,
                      const float* __restrict__ b2, float* __restrict__ out) {
    __shared__ float sg[EMBC], shid[EMBC];
    int bidx = blockIdx.x, j = threadIdx.x;
    sg[j] = g[(size_t)bidx * EMBC + j];
    __syncthreads();
    float acc = b1[j];
    for (int k = 0; k < EMBC; ++k) acc += sg[k] * W1[k * EMBC + j];
    shid[j] = fmaxf(acc, 0.0f);
    __syncthreads();
    if (j < NUM_CLASSESC) {
        float o = b2[j];
        for (int k = 0; k < EMBC; ++k) o += shid[k] * W2[k * NUM_CLASSESC + j];
        out[bidx * NUM_CLASSESC + j] = o;
    }
}

// ---------------------------------------------------------------- launch
extern "C" void kernel_launch(void* const* d_in, const int* in_sizes, int n_in,
                              void* d_out, int out_size, void* d_ws, size_t ws_size,
                              hipStream_t stream) {
    (void)in_sizes; (void)n_in; (void)out_size; (void)ws_size;

    const float* x            = (const float*)d_in[0];
    const int*   path_index   = (const int*)d_in[1];
    const int*   mask_index   = (const int*)d_in[2];
    const int*   path_lengths = (const int*)d_in[3];
    const int*   distances    = (const int*)d_in[4];
    const float* nmask        = (const float*)d_in[5];
    const int*   batch        = (const int*)d_in[6];
    const float* enc_W        = (const float*)d_in[7];
    const float* enc_b        = (const float*)d_in[8];
    const float* dist_emb     = (const float*)d_in[9];
    const float* Wih          = (const float*)d_in[10];
    const float* Whh          = (const float*)d_in[11];
    const float* bih          = (const float*)d_in[12];
    const float* bhh          = (const float*)d_in[13];
    const float* bn_gamma     = (const float*)d_in[14];
    const float* bn_beta      = (const float*)d_in[15];
    const float* mlp_W1       = (const float*)d_in[16];
    const float* mlp_b1       = (const float*)d_in[17];
    const float* mlp_W2       = (const float*)d_in[18];
    const float* mlp_b2       = (const float*)d_in[19];
    float* out = (float*)d_out;

    char* ws = (char*)d_ws;
    size_t off = 0;
    auto carve = [&](size_t bytes) -> char* {
        char* p = ws + off;
        off += (bytes + 255) & ~(size_t)255;
        return p;
    };
    float*    h       = (float*)   carve((size_t)N_NODESC * EMBC * 4);   // doubles as z
    float*    hn      = (float*)   carve((size_t)N_NODESC * EMBC * 4);
    _Float16* hn_h    = (_Float16*)carve((size_t)N_NODESC * EMBC * 2);
    _Float16* hs_h    = (_Float16*)carve((size_t)N_PATHSC * EMBC * 2);
    float*    cs      = (float*)   carve((size_t)N_PATHSC * EMBC * 4);
    float*    last    = (float*)   carve((size_t)N_PATHSC * EMBC * 4);
    _Float16* Wih_h   = (_Float16*)carve((size_t)NR_LSTMSC * GATESC * 256 * 2);
    float*    WihLast = (float*)   carve((size_t)NR_LSTMSC * GATESC * 4);
    _Float16* Whh_h   = (_Float16*)carve((size_t)NR_LSTMSC * GATESC * EMBC * 2);
    _Float16* dist_h  = (_Float16*)carve((size_t)(PATH_LENC + 2) * EMBC * 2);
    float*    stats   = (float*)   carve(2 * EMBC * 4);
    float*    gbuf    = (float*)   carve((size_t)NUM_GRAPHSC * EMBC * 4);

    k_prep<<<(NR_LSTMSC * GATESC * 257 + 255) / 256, 256, 0, stream>>>(
        Wih, Whh, dist_emb, Wih_h, WihLast, Whh_h, dist_h);
    k_encoder<<<N_NODESC, EMBC, 0, stream>>>(x, enc_W, enc_b, h);

    for (int l = 0; l < NR_LSTMSC; ++l) {
        k_normalize<<<N_NODESC, EMBC, 0, stream>>>(h, hn, hn_h);
        for (int t = 0; t < LSEQ; ++t) {
            k_lstm_step<<<N_PATHSC / 16, 256, 0, stream>>>(
                t, (t == 0) ? 1 : 0,
                path_index, distances, nmask, path_lengths,
                hn_h, dist_h,
                Wih_h + (size_t)l * GATESC * 256, WihLast + l * GATESC,
                Whh_h + (size_t)l * GATESC * EMBC,
                bih + l * GATESC, bhh + l * GATESC,
                hs_h, cs, last);
        }
        k_zero<<<(N_NODESC * EMBC + 255) / 256, 256, 0, stream>>>(h, N_NODESC * EMBC);
        k_scatter_paths<<<(N_PATHSC * EMBC + 255) / 256, 256, 0, stream>>>(last, mask_index, h);
        k_bn_stats<<<EMBC, 256, 0, stream>>>(h, hn, stats);
        k_bn_apply<<<(N_NODESC * EMBC + 255) / 256, 256, 0, stream>>>(
            h, hn, stats, bn_gamma + l * EMBC, bn_beta + l * EMBC);
    }

    k_zero<<<(NUM_GRAPHSC * EMBC + 255) / 256, 256, 0, stream>>>(gbuf, NUM_GRAPHSC * EMBC);
    k_scatter_graph<<<(N_NODESC * EMBC + 255) / 256, 256, 0, stream>>>(h, batch, gbuf);
    k_mlp<<<NUM_GRAPHSC, EMBC, 0, stream>>>(gbuf, mlp_W1, mlp_b1, mlp_W2, mlp_b2, out);
}
